// SwitchLayer_45105746543107
// MI455X (gfx1250) — matrix operand order repaired and checked by reading, not compiled, not run on hardware
//
#include <hip/hip_runtime.h>
#include <hip/hip_bf16.h>

// ---------- types for WMMA fragments (wave32, gfx1250) ----------
typedef __attribute__((ext_vector_type(16))) __bf16 v16bf;
typedef __attribute__((ext_vector_type(8)))  float  v8f;
typedef __attribute__((ext_vector_type(4)))  int    v4i;

union FragBF {
    v16bf v;
    uint4 q[2];
};

// float -> bf16 round-to-nearest-even (bit trick)
__device__ __forceinline__ unsigned short f2bf(float f) {
    unsigned u = __float_as_uint(f);
    unsigned r = u + 0x7FFFu + ((u >> 16) & 1u);
    return (unsigned short)(r >> 16);
}

__device__ __forceinline__ float gelu_exact(float u) {
    return 0.5f * u * (1.0f + erff(u * 0.70710678118654752f));
}

// ---------------- constants ----------------
#define NT     4096   // B*T tokens
#define DMODEL 512
#define DFF    2048
#define NE     8
#define MT     64     // tokens per block tile
#define FC     64     // dff chunk per iteration

// LDS pitches (ushort elements); every row 16B-aligned
#define PX   520      // sX  [64][512]
#define PW1  136      // sW1 [64][128] (f-major, k-minor)
#define PW2  40       // sW2 [512][32] (n-major, k-minor)
#define PH   72       // sH  [64][64]

// ---------- async global->LDS copy (gfx1250), with safe fallback ----------
#if __has_builtin(__builtin_amdgcn_global_load_async_to_lds_b128) && \
    __has_builtin(__builtin_amdgcn_s_wait_asynccnt)
#define HAVE_ASYNC_LDS 1
typedef __attribute__((address_space(1))) v4i gas_v4i;   // global
typedef __attribute__((address_space(3))) v4i las_v4i;   // LDS
#endif

__device__ __forceinline__ void cp16(const void* g, void* l) {
#ifdef HAVE_ASYNC_LDS
    __builtin_amdgcn_global_load_async_to_lds_b128(
        (gas_v4i*)(uintptr_t)g, (las_v4i*)(uintptr_t)l, 0, 0);
#else
    *(uint4*)l = *(const uint4*)g;
#endif
}
__device__ __forceinline__ void cp_wait() {
#ifdef HAVE_ASYNC_LDS
    __builtin_amdgcn_s_wait_asynccnt(0);
#endif
}

// ================= Prep A: fp32 [R][C] -> bf16 [C][R] tile transpose =================
__global__ __launch_bounds__(256)
void transpose_bf16(const float* __restrict__ src, unsigned short* __restrict__ dst,
                    int R, int C) {
    __shared__ float tile[32][33];
    const int ex = blockIdx.z;
    src += (size_t)ex * R * C;
    dst += (size_t)ex * R * C;
    const int c0 = blockIdx.x * 32, r0 = blockIdx.y * 32;
    const int tx = threadIdx.x & 31, ty = threadIdx.x >> 5;   // 32x8
#pragma unroll
    for (int i = ty; i < 32; i += 8)
        tile[i][tx] = src[(size_t)(r0 + i) * C + c0 + tx];
    __syncthreads();
#pragma unroll
    for (int i = ty; i < 32; i += 8)
        dst[(size_t)(c0 + i) * R + r0 + tx] = f2bf(tile[tx][i]);
}

// ================= Prep B: elementwise fp32 -> bf16 =================
__global__ __launch_bounds__(256)
void convert_bf16(const float* __restrict__ src, unsigned short* __restrict__ dst, int n) {
    int i = blockIdx.x * 256 + threadIdx.x;
    if (i < n) dst[i] = f2bf(src[i]);
}

// ================= Kernel 1: gating =================
// renormalized top-2 softmax weights == 1/(1+exp(m2-m1)) and complement
__global__ __launch_bounds__(256)
void moe_gate(const float* __restrict__ x, const float* __restrict__ Wg,
              float2* __restrict__ scales) {
    const int wave = threadIdx.x >> 5;
    const int lane = threadIdx.x & 31;
    const int t = blockIdx.x * 8 + wave;
    if (t >= NT) return;

    float acc[NE];
#pragma unroll
    for (int e = 0; e < NE; ++e) acc[e] = 0.f;

    for (int r = lane; r < DMODEL; r += 32) {
        float xv = x[(size_t)t * DMODEL + r];
#pragma unroll
        for (int e = 0; e < NE; ++e) acc[e] += xv * Wg[r * NE + e];
    }
#pragma unroll
    for (int e = 0; e < NE; ++e) {
#pragma unroll
        for (int off = 16; off > 0; off >>= 1)
            acc[e] += __shfl_xor(acc[e], off, 32);
    }
    float m1 = -1e30f, m2 = -1e30f;
#pragma unroll
    for (int e = 0; e < NE; ++e) {
        float v = acc[e];
        if (v > m1) { m2 = m1; m1 = v; }
        else if (v > m2) { m2 = v; }
    }
    float r = __expf(m2 - m1);
    float inv = 1.0f / (1.0f + r);
    if (lane == 0) scales[t] = make_float2(inv, r * inv);
}

// ================= Kernel 2: fused MoE FFN (all-bf16 staged) =================
__global__ __launch_bounds__(512)
void moe_main(const unsigned short* __restrict__ Xb,     // [NT][512] bf16
              const unsigned short* __restrict__ W1T,    // [E][2048 f][512 k] bf16
              const unsigned short* __restrict__ W2T,    // [E][512 n][2048 k] bf16
              const float* __restrict__ b1, const float* __restrict__ b2,
              const float2* __restrict__ scales, float* __restrict__ out) {
    __shared__ __align__(16) unsigned short sX[MT * PX];
    __shared__ __align__(16) unsigned short sW1[FC * PW1];
    __shared__ __align__(16) unsigned short sW2[DMODEL * PW2];
    __shared__ __align__(16) unsigned short sH[MT * PH];
    __shared__ float sS0[MT], sS1[MT];

    const int tid  = threadIdx.x;
    const int w    = tid >> 5;
    const int lane = tid & 31;
    const int lsub = lane & 15;
    const int hi   = lane >> 4;
    const int mfrag = w >> 2;           // token rows mfrag*16
    const int nq    = w & 3;            // GEMM1: 16-col frag; GEMM2: 128-col slice
    const int t0 = blockIdx.x * MT;

    // ---- stage X tile (pure bf16 copy) + gate scales ----
    for (int idx = tid; idx < MT * 64; idx += 512) {   // 4096 x 16B chunks
        int r = idx >> 6, c = (idx & 63) * 8;
        cp16(&Xb[(size_t)(t0 + r) * DMODEL + c], &sX[r * PX + c]);
    }
    if (tid < MT) {
        float2 s = scales[t0 + tid];
        sS0[tid] = s.x; sS1[tid] = s.y;
    }
    cp_wait();

    v8f Cout[8];
#pragma unroll
    for (int j = 0; j < 8; ++j)
        Cout[j] = (v8f){0.f, 0.f, 0.f, 0.f, 0.f, 0.f, 0.f, 0.f};

    for (int e = 0; e < NE; ++e) {
        const unsigned short* W1e = W1T + (size_t)e * DFF * DMODEL;   // [f][k]
        const unsigned short* W2e = W2T + (size_t)e * DMODEL * DFF;   // [n][k]

        for (int fc = 0; fc < DFF; fc += FC) {
            if (fc + FC < DFF) {
                __builtin_prefetch(&W1e[(size_t)(fc + FC) * DMODEL], 0, 1);
                __builtin_prefetch(&W2e[fc + FC], 0, 1);
            }

            // ---------- GEMM1: Y[64, FC] = X[64,512] @ W1[:, fc:fc+FC] ----------
            v8f yc = (v8f){0.f, 0.f, 0.f, 0.f, 0.f, 0.f, 0.f, 0.f};
            for (int ks = 0; ks < DMODEL; ks += 128) {
                __syncthreads();   // previous consumers of sW1 done
                for (int idx = tid; idx < FC * 16; idx += 512) {   // 1024 x 16B
                    int f = idx >> 4, c = (idx & 15) * 8;
                    cp16(&W1e[(size_t)(fc + f) * DMODEL + ks + c], &sW1[f * PW1 + c]);
                }
                cp_wait();
                __syncthreads();
#pragma unroll
                for (int k0 = 0; k0 < 128; k0 += 32) {
                    FragBF A;   // A 16x32 bf16 layout
                    const unsigned short* ax =
                        &sX[(mfrag * 16 + lsub) * PX + ks + k0 + hi * 8];
                    A.q[0] = *(const uint4*)(ax);
                    A.q[1] = *(const uint4*)(ax + 16);
                    FragBF B;   // B 32x16 bf16 layout
                    const unsigned short* bw =
                        &sW1[(nq * 16 + lsub) * PW1 + k0 + hi * 8];
                    B.q[0] = *(const uint4*)(bw);
                    B.q[1] = *(const uint4*)(bw + 16);
                    yc = __builtin_amdgcn_wmma_f32_16x16x32_bf16(
                        false, A.v, false, B.v, (short)0, yc, false, false);
                }
            }

            // ---------- activation: H = gelu(s0*Y+b1) + gelu(s1*Y+b1) ----------
            {
                const int colL = nq * 16 + lsub;
                const float b1v = b1[e * DFF + fc + colL];
#pragma unroll
                for (int v = 0; v < 8; ++v) {
                    int rloc = mfrag * 16 + hi * 8 + v;
                    float y = yc[v];
                    float hsum = gelu_exact(sS0[rloc] * y + b1v) +
                                 gelu_exact(sS1[rloc] * y + b1v);
                    sH[rloc * PH + colL] = f2bf(hsum);
                }
            }
            __syncthreads();

            // ---------- GEMM2: Out[64,512] += H[64,FC] @ W2[fc:fc+FC, :] ----------
            for (int ks2 = 0; ks2 < FC; ks2 += 32) {
                for (int idx = tid; idx < DMODEL * 4; idx += 512) {  // 2048 x 16B
                    int n = idx >> 2, c = (idx & 3) * 8;
                    cp16(&W2e[(size_t)n * DFF + fc + ks2 + c], &sW2[n * PW2 + c]);
                }
                cp_wait();
                __syncthreads();
                FragBF A;
                const unsigned short* ah =
                    &sH[(mfrag * 16 + lsub) * PH + ks2 + hi * 8];
                A.q[0] = *(const uint4*)(ah);
                A.q[1] = *(const uint4*)(ah + 16);
#pragma unroll
                for (int j = 0; j < 8; ++j) {
                    FragBF B;
                    const unsigned short* bw =
                        &sW2[(nq * 128 + j * 16 + lsub) * PW2 + hi * 8];
                    B.q[0] = *(const uint4*)(bw);
                    B.q[1] = *(const uint4*)(bw + 16);
                    Cout[j] = __builtin_amdgcn_wmma_f32_16x16x32_bf16(
                        false, A.v, false, B.v, (short)0, Cout[j], false, false);
                }
                __syncthreads();
            }
        }
    }

    // ---------- epilogue: out = Cout + K * sum_e b2[e]  (K = 2) ----------
#pragma unroll
    for (int j = 0; j < 8; ++j) {
        const int col = nq * 128 + j * 16 + lsub;
        float bsum = 0.f;
#pragma unroll
        for (int e = 0; e < NE; ++e) bsum += b2[e * DMODEL + col];
#pragma unroll
        for (int v = 0; v < 8; ++v) {
            int row = t0 + mfrag * 16 + hi * 8 + v;
            out[(size_t)row * DMODEL + col] = Cout[j][v] + 2.0f * bsum;
        }
    }
}

// ================= host launcher =================
extern "C" void kernel_launch(void* const* d_in, const int* in_sizes, int n_in,
                              void* d_out, int out_size, void* d_ws, size_t ws_size,
                              hipStream_t stream) {
    (void)in_sizes; (void)n_in; (void)out_size; (void)ws_size;
    const float* x  = (const float*)d_in[0];
    const float* Wg = (const float*)d_in[1];
    const float* W1 = (const float*)d_in[2];
    const float* b1 = (const float*)d_in[3];
    const float* W2 = (const float*)d_in[4];
    const float* b2 = (const float*)d_in[5];
    float* out = (float*)d_out;

    // workspace layout (bf16 elements)
    unsigned short* W1T = (unsigned short*)d_ws;              // 8*2048*512
    unsigned short* W2T = W1T + (size_t)NE * DFF * DMODEL;    // 8*512*2048
    unsigned short* Xb  = W2T + (size_t)NE * DMODEL * DFF;    // 4096*512
    float2* scales = (float2*)(Xb + (size_t)NT * DMODEL);     // 4096

    // W1[e]: fp32 [512 k][2048 f] -> bf16 [2048 f][512 k]
    transpose_bf16<<<dim3(DFF / 32, DMODEL / 32, NE), 256, 0, stream>>>(W1, W1T, DMODEL, DFF);
    // W2[e]: fp32 [2048 k][512 n] -> bf16 [512 n][2048 k]
    transpose_bf16<<<dim3(DMODEL / 32, DFF / 32, NE), 256, 0, stream>>>(W2, W2T, DFF, DMODEL);
    // X: fp32 -> bf16 (same layout)
    convert_bf16<<<(NT * DMODEL + 255) / 256, 256, 0, stream>>>(x, Xb, NT * DMODEL);

    moe_gate<<<NT / 8, 256, 0, stream>>>(x, Wg, scales);
    moe_main<<<NT / MT, 512, 0, stream>>>(Xb, W1T, W2T, b1, b2, scales, out);
}
